// LSTM_Encoder_80865644249211
// MI455X (gfx1250) — matrix operand order repaired and checked by reading, not compiled, not run on hardware
//
#include <hip/hip_runtime.h>
#include <hip/hip_bf16.h>
#include <math.h>

// ---- problem constants (from reference) ----
#define BB 16
#define LL 128
#define IN_DIM 1218
#define K0PAD 1248          // 1218 padded to 32
#define HID 200             // H per direction
#define G4 800              // 4*H
#define LSTM_DIM 400
#define K1PAD 416           // 400 padded to 32
#define KHPAD 224           // 200 padded to 32
#define SMAX 8001           // (L-1)(L-2)/2
#define SPAN_ELEMS ((size_t)BB * SMAX * LSTM_DIM)

typedef __bf16 bf16_t;
typedef __attribute__((ext_vector_type(16))) __bf16 v16bf;
typedef __attribute__((ext_vector_type(8)))  float  v8f;

union Frag16 { v16bf v; uint4 q[2]; };

__device__ __forceinline__ float sigm(float x) { return 1.0f / (1.0f + __expf(-x)); }

// CDNA5 async global->LDS copy (ASYNCcnt-tracked). Low 32 bits of a generic
// pointer to __shared__ are the LDS byte offset on AMDGPU.
__device__ __forceinline__ void async_b128(unsigned lds_off, const float* gsrc) {
    asm volatile("global_load_async_to_lds_b128 %0, %1, off"
                 :: "v"(lds_off), "v"(gsrc)
                 : "memory");
}
__device__ __forceinline__ void wait_async0() {
    asm volatile("s_wait_asynccnt 0x0" ::: "memory");
}

// ---------------- lengths + second output ----------------
__global__ void lens_kernel(const int* __restrict__ word_idxs, int* __restrict__ lens,
                            float* __restrict__ d_out) {
    int b = threadIdx.x;
    if (b < BB) {
        int cnt = 0;
        for (int l = 0; l < LL; ++l) cnt += (word_idxs[b * LL + l] != 0);
        lens[b] = cnt;
        d_out[SPAN_ELEMS + b] = (float)(cnt - 2);
    }
}

// ---------------- embedding gather + concat -> bf16 (padded) ----------------
__global__ void embed_kernel(const float* __restrict__ ext_emb, const float* __restrict__ word_emb,
                             const float* __restrict__ pos_emb, const float* __restrict__ dep_emb,
                             const float* __restrict__ ent_emb, const float* __restrict__ iob_emb,
                             const float* __restrict__ bert,
                             const int* __restrict__ wi_arr, const int* __restrict__ pos_i,
                             const int* __restrict__ dep_i, const int* __restrict__ ent_i,
                             const int* __restrict__ iob_i,
                             bf16_t* __restrict__ X0) {
    int r = blockIdx.x;                 // r = b*L + l
    int wi  = wi_arr[r];
    int wid = (wi >= 20000) ? 1 : wi;
    int pi = pos_i[r], di = dep_i[r], ei = ent_i[r], ii = iob_i[r];
    for (int d = threadIdx.x; d < K0PAD; d += blockDim.x) {
        float v;
        if      (d < 300)  v = ext_emb[(size_t)wi * 300 + d] + word_emb[(size_t)wid * 300 + d];
        else if (d < 350)  v = pos_emb[pi * 50 + (d - 300)];
        else if (d < 400)  v = dep_emb[di * 50 + (d - 350)];
        else if (d < 425)  v = ent_emb[ei * 25 + (d - 400)];
        else if (d < 450)  v = iob_emb[ii * 25 + (d - 425)];
        else if (d < 1218) v = bert[(size_t)r * 768 + (d - 450)];
        else               v = 0.0f;
        X0[(size_t)r * K0PAD + d] = (bf16_t)v;
    }
}

// ---------------- f32 -> bf16 with K padding ----------------
__global__ void convert_pad(const float* __restrict__ src, bf16_t* __restrict__ dst,
                            int K, int Kpad, int total) {
    int idx = blockIdx.x * blockDim.x + threadIdx.x;
    if (idx >= total) return;
    int row = idx / Kpad;
    int k   = idx - row * Kpad;
    dst[idx] = (k < K) ? (bf16_t)src[(size_t)row * K + k] : (bf16_t)0.0f;
}

// ---------------- gx = X @ Wih^T + bih + bhh  (both dirs), WMMA bf16 ----------------
// X: (2048, Kpad) bf16 row-major. W: (2, 800, Kpad) bf16. gx out: (2, L, B, 800) f32.
__global__ __launch_bounds__(256)
void gemm_gx(const bf16_t* __restrict__ X, const bf16_t* __restrict__ W,
             const float* __restrict__ bih, const float* __restrict__ bhh,
             float* __restrict__ gx, int ksteps, int ldx) {
    int wid  = blockIdx.x * (blockDim.x >> 5) + (threadIdx.x >> 5);
    int lane = threadIdx.x & 31;
    int dir  = wid / 6400;              // 128 Mtiles * 50 Ntiles
    int rem  = wid - dir * 6400;
    int mt   = rem / 50;
    int nt   = rem - mt * 50;

    int col = lane & 15;
    int aHi = (lane >> 4) * 8;          // K sub-offset for A lanes 16..31
    int bHi = (lane >> 4) * 8;          // K sub-offset for B lanes 16..31
    int m   = mt * 16 + col;            // X row (= b*L + l)
    int n   = nt * 16 + col;            // gate column g

    const bf16_t* Arow = X + (size_t)m * ldx;
    const bf16_t* Brow = W + ((size_t)dir * G4 + n) * ldx;

    v8f c = {};
    for (int ks = 0; ks < ksteps; ++ks) {
        int kb = ks * 32;
        Frag16 a, bfr;
        a.q[0]   = *(const uint4*)(Arow + kb + aHi);
        a.q[1]   = *(const uint4*)(Arow + kb + aHi + 16);
        bfr.q[0] = *(const uint4*)(Brow + kb + bHi);
        bfr.q[1] = *(const uint4*)(Brow + kb + bHi + 16);
        __builtin_prefetch(Brow + kb + 64, 0, 1);   // weight tile two k-steps ahead -> global_prefetch_b8
        c = __builtin_amdgcn_wmma_f32_16x16x32_bf16(false, a.v, false, bfr.v,
                                                    (short)0, c, false, false);
    }
    float bias = bih[dir * G4 + n] + bhh[dir * G4 + n];
    #pragma unroll
    for (int r = 0; r < 8; ++r) {
        int row = mt * 16 + r + aHi;    // = b*L + l
        int b   = row >> 7;
        int l   = row & 127;
        gx[(((size_t)dir * LL + l) * BB + b) * G4 + n] = c[r] + bias;
    }
}

// ---------------- one BiLSTM layer direction: LDS-resident recurrence ----------------
// gx: (2, L, B, 800) f32. Whh: (2, 800, 224) bf16. hout: (B, L, 400) f32.
// The next timestep's gx row (51.2 KB) is staged into LDS with
// global_load_async_to_lds_b128 (ASYNCcnt) while the current step computes.
__global__ __launch_bounds__(512)
void lstm_layer(const float* __restrict__ gx, const bf16_t* __restrict__ Whh,
                const int* __restrict__ lens, float* __restrict__ hout) {
    const int dir = blockIdx.x;         // 0 = fwd, 1 = rev
    __shared__ float  gxs[2][BB][G4];   // ping-pong gx stage; g written back in place
    __shared__ bf16_t hA[16][KHPAD];    // h state as bf16 A-fragments (K padded)
    __shared__ float  hS[16][HID];
    __shared__ float  cS[16][HID];

    for (int idx = threadIdx.x; idx < 16 * KHPAD; idx += 512) hA[idx / KHPAD][idx % KHPAD] = (bf16_t)0.0f;
    for (int idx = threadIdx.x; idx < 16 * HID; idx += 512) { hS[idx / HID][idx % HID] = 0.0f; cS[idx / HID][idx % HID] = 0.0f; }

    const float* gx_dir = gx + ((size_t)dir * LL) * BB * G4;
    auto issue_stage = [&](int buf, int pos) {
        const float* src = gx_dir + (size_t)pos * BB * G4;
        unsigned base = (unsigned)(uintptr_t)(&gxs[buf][0][0]);
        for (int idx = threadIdx.x; idx < (BB * G4) / 4; idx += 512)
            async_b128(base + idx * 16, src + idx * 4);
    };
    issue_stage(0, dir ? (LL - 1) : 0);
    __syncthreads();

    int lane = threadIdx.x & 31;
    int wv   = threadIdx.x >> 5;        // 0..15
    int col  = lane & 15;
    int aHi  = (lane >> 4) * 8;
    int bHi  = (lane >> 4) * 8;

    for (int t = 0; t < LL; ++t) {
        int pos = dir ? (LL - 1 - t) : t;
        int cur = t & 1;

        // current gx stage complete (own wave), then visible to all waves
        wait_async0();
        __syncthreads();

        // kick off async stage of next timestep's gx row
        if (t + 1 < LL) issue_stage(cur ^ 1, dir ? (LL - 2 - t) : (t + 1));

        // ---- g = gx + h @ Whh^T via WMMA; write g back into gxs[cur] in place ----
        for (int nt = wv; nt < 50; nt += 16) {
            int n = nt * 16 + col;
            const bf16_t* Brow = Whh + ((size_t)dir * G4 + n) * KHPAD;
            v8f c = {};
            #pragma unroll
            for (int ks = 0; ks < 7; ++ks) {
                int kb = ks * 32;
                Frag16 a, bfr;
                a.q[0]   = *(const uint4*)(&hA[col][kb + aHi]);
                a.q[1]   = *(const uint4*)(&hA[col][kb + aHi + 16]);
                bfr.q[0] = *(const uint4*)(Brow + kb + bHi);
                bfr.q[1] = *(const uint4*)(Brow + kb + bHi + 16);
                c = __builtin_amdgcn_wmma_f32_16x16x32_bf16(false, a.v, false, bfr.v,
                                                            (short)0, c, false, false);
            }
            #pragma unroll
            for (int r = 0; r < 8; ++r) {
                int b = r + aHi;
                gxs[cur][b][n] = c[r] + gxs[cur][b][n];
            }
        }
        __syncthreads();

        // ---- gates + state update ----
        for (int idx = threadIdx.x; idx < BB * HID; idx += 512) {
            int b  = idx / HID;
            int hi = idx - b * HID;
            float gi = gxs[cur][b][hi];
            float gf = gxs[cur][b][HID + hi];
            float gg = gxs[cur][b][2 * HID + hi];
            float go = gxs[cur][b][3 * HID + hi];
            float cO = cS[b][hi], hO = hS[b][hi];
            float cn = sigm(gf) * cO + sigm(gi) * tanhf(gg);
            float hn = sigm(go) * tanhf(cn);
            bool  mt = pos < lens[b];
            float hN = mt ? hn : hO;
            float cN = mt ? cn : cO;
            hS[b][hi] = hN;
            cS[b][hi] = cN;
            hA[b][hi] = (bf16_t)hN;
            hout[((size_t)b * LL + pos) * LSTM_DIM + dir * HID + hi] = mt ? hn : 0.0f;
        }
        __syncthreads();
    }
}

// ---------------- span features: closed-form compaction, float4 stores ----------------
// h1: (B, L, 400) f32. out row k of batch b = k-th (i,j), i<j<=len-2, row-major.
__global__ __launch_bounds__(128)
void span_kernel(const float* __restrict__ h1, const int* __restrict__ lens,
                 float* __restrict__ out) {
    int b = blockIdx.x / SMAX;
    int k = blockIdx.x - b * SMAX;
    int tid = threadIdx.x;
    if (tid >= 100) return;
    int d4 = tid * 4;
    float4* dst = (float4*)(out + ((size_t)b * SMAX + k) * LSTM_DIM + d4);

    int n = lens[b] - 1;                     // #positions with m1 true
    int total = (n * (n - 1)) >> 1;
    if (k >= total) { *dst = make_float4(0.f, 0.f, 0.f, 0.f); return; }

    // invert offset(i) = i*(n-1) - i*(i-1)/2  <= k
    float A = (float)(2 * n - 1);
    int i = (int)((A - sqrtf(fmaxf(A * A - 8.0f * (float)k, 0.0f))) * 0.5f);
    if (i < 0) i = 0;
    if (i > n - 2) i = n - 2;
    #define OFF(x) ((x) * (n - 1) - ((x) * ((x) - 1)) / 2)
    while (i > 0 && OFF(i) > k) --i;
    while (i < n - 2 && OFF(i + 1) <= k) ++i;
    int j = i + 1 + (k - OFF(i));
    #undef OFF

    const float* base = h1 + (size_t)b * LL * LSTM_DIM;
    float4 rJ, rI;
    if (tid < 50) {      // forward half: hf[j] - hf[i], row offset d4 in [0,200)
        rJ = *(const float4*)(base + (size_t)j * LSTM_DIM + d4);
        rI = *(const float4*)(base + (size_t)i * LSTM_DIM + d4);
        *dst = make_float4(rJ.x - rI.x, rJ.y - rI.y, rJ.z - rI.z, rJ.w - rI.w);
    } else {             // backward half: hb[i+1] - hb[j+1], row offset d4 in [200,400)
        rI = *(const float4*)(base + (size_t)(i + 1) * LSTM_DIM + d4);
        rJ = *(const float4*)(base + (size_t)(j + 1) * LSTM_DIM + d4);
        *dst = make_float4(rI.x - rJ.x, rI.y - rJ.y, rI.z - rJ.z, rI.w - rJ.w);
    }
}

// ---------------- host launcher ----------------
extern "C" void kernel_launch(void* const* d_in, const int* in_sizes, int n_in,
                              void* d_out, int out_size, void* d_ws, size_t ws_size,
                              hipStream_t stream) {
    const float* ext_emb  = (const float*)d_in[0];
    const float* word_emb = (const float*)d_in[1];
    const float* pos_emb  = (const float*)d_in[2];
    const float* dep_emb  = (const float*)d_in[3];
    const float* ent_emb  = (const float*)d_in[4];
    const float* iob_emb  = (const float*)d_in[5];
    const float* w_ih_l0  = (const float*)d_in[6];
    const float* w_hh_l0  = (const float*)d_in[7];
    const float* b_ih_l0  = (const float*)d_in[8];
    const float* b_hh_l0  = (const float*)d_in[9];
    const float* w_ih_l1  = (const float*)d_in[10];
    const float* w_hh_l1  = (const float*)d_in[11];
    const float* b_ih_l1  = (const float*)d_in[12];
    const float* b_hh_l1  = (const float*)d_in[13];
    const float* bert     = (const float*)d_in[14];
    const int*   word_i   = (const int*)d_in[15];
    const int*   pos_i    = (const int*)d_in[16];
    const int*   dep_i    = (const int*)d_in[17];
    const int*   ent_i    = (const int*)d_in[18];
    const int*   iob_i    = (const int*)d_in[19];
    float* out = (float*)d_out;

    // workspace carve-up (256B aligned)
    char* ws = (char*)d_ws;
    size_t off = 0;
    auto alloc = [&](size_t bytes) { size_t r = off; off = (off + bytes + 255) & ~(size_t)255; return r; };
    size_t oX0   = alloc((size_t)2048 * K0PAD * 2);
    size_t oWih0 = alloc((size_t)2 * G4 * K0PAD * 2);
    size_t oWhh0 = alloc((size_t)2 * G4 * KHPAD * 2);
    size_t oWih1 = alloc((size_t)2 * G4 * K1PAD * 2);
    size_t oWhh1 = alloc((size_t)2 * G4 * KHPAD * 2);
    size_t oGX   = alloc((size_t)2 * LL * BB * G4 * 4);
    size_t oH0   = alloc((size_t)BB * LL * LSTM_DIM * 4);
    size_t oX1   = alloc((size_t)2048 * K1PAD * 2);
    size_t oH1   = alloc((size_t)BB * LL * LSTM_DIM * 4);
    size_t oLEN  = alloc((size_t)BB * 4);
    if (off > ws_size) return;

    bf16_t* X0   = (bf16_t*)(ws + oX0);
    bf16_t* Wih0 = (bf16_t*)(ws + oWih0);
    bf16_t* Whh0 = (bf16_t*)(ws + oWhh0);
    bf16_t* Wih1 = (bf16_t*)(ws + oWih1);
    bf16_t* Whh1 = (bf16_t*)(ws + oWhh1);
    float*  GX   = (float*)(ws + oGX);
    float*  H0   = (float*)(ws + oH0);
    bf16_t* X1   = (bf16_t*)(ws + oX1);
    float*  H1   = (float*)(ws + oH1);
    int*    LENS = (int*)(ws + oLEN);

    // 1. lengths + tail output
    lens_kernel<<<1, 32, 0, stream>>>(word_i, LENS, out);

    // 2. embeddings -> X0 bf16 (padded)
    embed_kernel<<<BB * LL, 256, 0, stream>>>(ext_emb, word_emb, pos_emb, dep_emb, ent_emb,
                                              iob_emb, bert, word_i, pos_i, dep_i, ent_i,
                                              iob_i, X0);

    // 3. weight conversion to bf16 (padded K)
    {
        int t0 = 2 * G4 * K0PAD; convert_pad<<<(t0 + 255) / 256, 256, 0, stream>>>(w_ih_l0, Wih0, IN_DIM, K0PAD, t0);
        int t1 = 2 * G4 * KHPAD; convert_pad<<<(t1 + 255) / 256, 256, 0, stream>>>(w_hh_l0, Whh0, HID, KHPAD, t1);
        int t2 = 2 * G4 * K1PAD; convert_pad<<<(t2 + 255) / 256, 256, 0, stream>>>(w_ih_l1, Wih1, LSTM_DIM, K1PAD, t2);
        int t3 = 2 * G4 * KHPAD; convert_pad<<<(t3 + 255) / 256, 256, 0, stream>>>(w_hh_l1, Whh1, HID, KHPAD, t3);
    }

    // 4. layer-0 input projection (WMMA): 12800 waves
    gemm_gx<<<1600, 256, 0, stream>>>(X0, Wih0, b_ih_l0, b_hh_l0, GX, K0PAD / 32, K0PAD);

    // 5. layer-0 recurrence (1 workgroup per direction, LDS-resident, async gx staging)
    lstm_layer<<<2, 512, 0, stream>>>(GX, Whh0, LENS, H0);

    // 6. h0 -> bf16 (padded) for layer-1 GEMM
    {
        int t = 2048 * K1PAD; convert_pad<<<(t + 255) / 256, 256, 0, stream>>>(H0, X1, LSTM_DIM, K1PAD, t);
    }

    // 7. layer-1 input projection (WMMA)
    gemm_gx<<<1600, 256, 0, stream>>>(X1, Wih1, b_ih_l1, b_hh_l1, GX, K1PAD / 32, K1PAD);

    // 8. layer-1 recurrence
    lstm_layer<<<2, 512, 0, stream>>>(GX, Whh1, LENS, H1);

    // 9. span compaction (HBM-store bound: ~205 MB of float4 stores)
    span_kernel<<<BB * SMAX, 128, 0, stream>>>(H1, LENS, out);
}
